// KANvolution_18262200943482
// MI455X (gfx1250) — matrix operand order repaired and checked by reading, not compiled, not run on hardware
//
#include <hip/hip_runtime.h>
#include <hip/hip_bf16.h>

// ---------------------------------------------------------------------------
// KANvolution on gfx1250:
//   features: 17 hat-basis values (<=2 nonzero) + SiLU  -> 18 per element
//   implicit GEMM: M=B*Ho*Wo=16384, N=F=64, K=C*KH*KW*18=5184 (bf16 WMMA)
//   GEMM: 64x64 tile / WG, K=64 per double-buffered LDS stage, 81 stages
// ---------------------------------------------------------------------------

typedef __attribute__((ext_vector_type(16))) __bf16 v16bf;
typedef __attribute__((ext_vector_type(8)))  float  v8f;

#define B_  4
#define H_  66
#define W_  66
#define C_  32
#define F_  64
#define HO_ 64
#define WO_ 64
#define G_  16            // GRID_SIZE
#define KPERX 18          // 17 basis + silu
#define KCH   576         // C_*KPERX  (contiguous K run per (i,j) tap)
#define KTOT  5184        // 9*KCH
#define NPIX  (B_*H_*W_)  // input pixels
#define NELEM (NPIX*C_)   // input elements
#define MTOT  (B_*HO_*WO_)
#define NSTAGE 81         // KTOT / 64

static __device__ __forceinline__ unsigned short f2bf(float f) {
    unsigned int u = __float_as_uint(f);
    unsigned int r = (u + 0x7FFFu + ((u >> 16) & 1u)) >> 16;   // RNE
    return (unsigned short)r;
}

// ---------------------------------------------------------------------------
// Kernel 1: featurize x -> feat[pixel][c*18 + q]  (bf16)
// ---------------------------------------------------------------------------
__global__ void kan_featurize(const float* __restrict__ x,
                              unsigned short* __restrict__ feat) {
    int idx = blockIdx.x * blockDim.x + threadIdx.x;
    if (idx >= NELEM) return;

    float v  = x[idx];
    float si = v / (1.0f + __expf(-v));                 // SiLU on unclipped x
    float xc = fminf(1.0f, fmaxf(-1.0f, v));
    float tp = (xc + 1.0f) * 8.0f;                      // / grid_spacing(0.125)
    int   g  = (int)floorf(tp);
    if (g > G_) g = G_;
    float fr  = tp - (float)g;
    float inv = 1.0f / (1.0f + 1e-8f);                  // sum(w)==1 interior

    unsigned short h[KPERX];
#pragma unroll
    for (int q = 0; q < KPERX; ++q) h[q] = 0;
    h[g] = f2bf((1.0f - fr) * inv);
    if (g < G_) h[g + 1] = f2bf(fr * inv);
    h[17] = f2bf(si);

    int pixel = idx / C_;
    int c     = idx - pixel * C_;
    unsigned int* dst =
        (unsigned int*)(feat + (size_t)pixel * KCH + (size_t)c * KPERX);
#pragma unroll
    for (int q = 0; q < 9; ++q)
        dst[q] = (unsigned int)h[2 * q] | ((unsigned int)h[2 * q + 1] << 16);
}

// ---------------------------------------------------------------------------
// Kernel 2: pack weights -> Wpack[f][k],  k = (i*3+j)*576 + c*18 + q
// ---------------------------------------------------------------------------
__global__ void kan_pack_w(const float* __restrict__ cp,
                           const float* __restrict__ wsp,
                           const float* __restrict__ wsl,
                           unsigned short* __restrict__ Wp) {
    int idx = blockIdx.x * blockDim.x + threadIdx.x;   // over F*C*3*3 = 18432
    if (idx >= F_ * C_ * 9) return;

    int j = idx % 3;
    int i = (idx / 3) % 3;
    int c = (idx / 9) % C_;
    int f = idx / (9 * C_);

    float ws = wsp[idx];
    float wl = wsl[idx];
    const float* cpe = cp + (size_t)idx * (G_ + 1);

    unsigned short h[KPERX];
#pragma unroll
    for (int q = 0; q <= G_; ++q) h[q] = f2bf(ws * cpe[q]);
    h[17] = f2bf(wl);

    size_t kglob = (size_t)(i * 3 + j) * KCH + (size_t)c * KPERX;
    unsigned int* dst = (unsigned int*)(Wp + (size_t)f * KTOT + kglob);
#pragma unroll
    for (int q = 0; q < 9; ++q)
        dst[q] = (unsigned int)h[2 * q] | ((unsigned int)h[2 * q + 1] << 16);
}

// ---------------------------------------------------------------------------
// Kernel 3: implicit-GEMM conv with v_wmma_f32_16x16x32_bf16
// ---------------------------------------------------------------------------
struct StageRegs { uint4 a0, a1, b0, b1; };

__device__ __forceinline__ void stage_load(const unsigned short* __restrict__ feat,
                                           const unsigned short* __restrict__ Wp,
                                           int s, int bb, int ho, int wo,
                                           int lr, int lc16, StageRegs& st) {
    const int ij = s / 9;
    const int t  = s - ij * 9;           // 0..8 -> K offset t*64 within the run
    const int i  = ij / 3;
    const int j  = ij - i * 3;
    const size_t arow =
        ((size_t)(bb * H_ + ho + i) * W_ + (wo + j)) * (size_t)KCH
        + (size_t)t * 64 + lc16;
    const size_t brow =
        (size_t)lr * KTOT + (size_t)ij * KCH + (size_t)t * 64 + lc16;
    st.a0 = *(const uint4*)(feat + arow);
    st.a1 = *(const uint4*)(feat + arow + 8);
    st.b0 = *(const uint4*)(Wp + brow);
    st.b1 = *(const uint4*)(Wp + brow + 8);
    __builtin_prefetch(feat + arow + 64, 0, 0);   // speculative: next K slab
}

__device__ __forceinline__ void stage_store(unsigned int* __restrict__ ldsA,
                                            unsigned int* __restrict__ ldsB,
                                            int lr, int lc16, const StageRegs& st) {
    unsigned int* pa = ldsA + lr * 32 + (lc16 >> 1);
    pa[0] = st.a0.x; pa[1] = st.a0.y; pa[2] = st.a0.z; pa[3] = st.a0.w;
    pa[4] = st.a1.x; pa[5] = st.a1.y; pa[6] = st.a1.z; pa[7] = st.a1.w;
    unsigned int* pb = ldsB + lr * 32 + (lc16 >> 1);
    pb[0] = st.b0.x; pb[1] = st.b0.y; pb[2] = st.b0.z; pb[3] = st.b0.w;
    pb[4] = st.b1.x; pb[5] = st.b1.y; pb[6] = st.b1.z; pb[7] = st.b1.w;
}

__global__ __launch_bounds__(256) void kan_gemm(
        const unsigned short* __restrict__ feat,
        const unsigned short* __restrict__ Wp,
        const float* __restrict__ bias,
        float* __restrict__ out) {
    // double-buffered 64x64 bf16 tiles: [buf][row][k-pair(u32)]
    __shared__ unsigned int ldsA[2][64 * 32];
    __shared__ unsigned int ldsB[2][64 * 32];

    const int tid  = threadIdx.x;
    const int lane = tid & 31;
    const int wave = tid >> 5;
    const int Mbase = blockIdx.x * 64;

    // staging coords: 64 rows x 64 cols bf16, 32B (2x b128) per thread/tile
    const int lr   = tid >> 2;          // 0..63 (A row / B row(=f))
    const int lc16 = (tid & 3) * 16;    // bf16 col offset 0,16,32,48

    // decompose this thread's A row -> (b, ho, wo)
    const int p  = Mbase + lr;
    const int wo = p & 63;
    const int ho = (p >> 6) & 63;
    const int bb = p >> 12;

    const int nsub  = wave & 3;
    const int mpair = wave >> 2;        // 0..1 -> m rows [mpair*32, mpair*32+32)
    const int half  = lane >> 4;
    const int l16   = lane & 15;

    v8f acc0 = {};
    v8f acc1 = {};

    // prologue: stage 0 into buffer 0
    StageRegs st;
    stage_load(feat, Wp, 0, bb, ho, wo, lr, lc16, st);
    stage_store(&ldsA[0][0], &ldsB[0][0], lr, lc16, st);

    for (int s = 0; s < NSTAGE; ++s) {
        __syncthreads();                        // stage s visible to all waves
        const int buf  = s & 1;
        const bool more = (s + 1 < NSTAGE);

        StageRegs nx;
        if (more) stage_load(feat, Wp, s + 1, bb, ho, wo, lr, lc16, nx);

        const unsigned int* A  = &ldsA[buf][0];
        const unsigned int* Bm = &ldsB[buf][0];

#pragma unroll
        for (int cc = 0; cc < 2; ++cc) {        // two K=32 chunks per stage
            union { unsigned int u[8]; v16bf v; } fa0, fa1, fb;
#pragma unroll
            for (int vv = 0; vv < 8; ++vv) {
                // A k order used here (same as B): VGPR v -> k=2v (+16*half)
                const int pA = cc * 16 + vv + 8 * half;
                fa0.u[vv] = A[(mpair * 32 +      l16) * 32 + pA];
                fa1.u[vv] = A[(mpair * 32 + 16 + l16) * 32 + pA];
                // B k order used here (same as A): VGPR v -> k=2v (+16*half)
                const int pB = cc * 16 + vv + 8 * half;
                fb.u[vv]  = Bm[(nsub * 16 + l16) * 32 + pB];
            }
            acc0 = __builtin_amdgcn_wmma_f32_16x16x32_bf16(
                       false, fa0.v, false, fb.v, (short)0, acc0, false, false);
            acc1 = __builtin_amdgcn_wmma_f32_16x16x32_bf16(
                       false, fa1.v, false, fb.v, (short)0, acc1, false, false);
        }

        if (more)
            stage_store(&ldsA[buf ^ 1][0], &ldsB[buf ^ 1][0], lr, lc16, nx);
    }

    // epilogue: C layout -> lane holds N=l16, M = r + 8*half per VGPR r
    const int n  = nsub * 16 + l16;
    const float bv = bias[n];
#pragma unroll
    for (int r = 0; r < 8; ++r) {
        const int m = mpair * 32 + r + 8 * half;
        out[(size_t)(Mbase + m)      * F_ + n] = acc0[r] + bv;
        out[(size_t)(Mbase + m + 16) * F_ + n] = acc1[r] + bv;
    }
}

// ---------------------------------------------------------------------------
extern "C" void kernel_launch(void* const* d_in, const int* in_sizes, int n_in,
                              void* d_out, int out_size, void* d_ws, size_t ws_size,
                              hipStream_t stream) {
    const float* x   = (const float*)d_in[0];
    const float* cp  = (const float*)d_in[1];  // (F,C,3,3,17)
    const float* wsp = (const float*)d_in[2];  // (F,C,3,3)
    const float* wsl = (const float*)d_in[3];  // (F,C,3,3)
    const float* bia = (const float*)d_in[4];  // (F,)
    float* out = (float*)d_out;                // (B,Ho,Wo,F) f32

    unsigned short* feat = (unsigned short*)d_ws;              // NPIX*KCH bf16
    unsigned short* Wp   = feat + (size_t)NPIX * KCH;          // F*KTOT bf16

    kan_featurize<<<(NELEM + 255) / 256, 256, 0, stream>>>(x, feat);
    kan_pack_w<<<(F_ * C_ * 9 + 255) / 256, 256, 0, stream>>>(cp, wsp, wsl, Wp);
    kan_gemm<<<MTOT / 64, 256, 0, stream>>>(feat, Wp, bia, out);
}
